// ButterflyGatingUnit_77214922047687
// MI455X (gfx1250) — matrix operand order repaired and checked by reading, not compiled, not run on hardware
//
#include <hip/hip_runtime.h>

typedef __bf16 bf16;
typedef __attribute__((ext_vector_type(16))) __bf16 v16bf;
typedef __attribute__((ext_vector_type(8)))  float  v8f;

// ---------------- problem constants ----------------
constexpr int B_  = 8, C_ = 96, H_ = 128, W_ = 128;
constexpr int HW  = H_ * W_;                        // 16384
constexpr long long NCHW = (long long)B_ * C_ * HW; // 12,582,912
constexpr int KTOT3 = C_ * 9;                       // 864 = 27*32
constexpr int KS3   = 27;
constexpr int KS1   = 6;                            // 192 = 6*32
constexpr int PK3_SZ = 6 * KS3 * 32 * 16;           // 82944
constexpr int PK1_SZ = 6 * KS1 * 32 * 16;           // 18432
constexpr int LPATCH = 42 * 42;                     // 1764
constexpr int AKS    = 56;                          // ceil(1764/32)
constexpr long long ASTRM = (long long)B_ * 9 * 6 * AKS * 512; // 12,386,304
constexpr float EPS_ = 1e-5f;
constexpr int CST = 104;   // conv LDS ci-stride (208B: 16B aligned, conflict-free)
constexpr int YST = 200;   // final LDS ci-stride (400B)

// A-fragment element -> K offset (contiguous: k = 16*(lane/16) + e, matches B loads)
__device__ __forceinline__ int koffA(int lane, int e) {
  return e + ((lane >> 4) << 4);
}

// 32-byte fragment load with only 16B alignment assumed (global or LDS)
__device__ __forceinline__ v16bf ldfrag(const bf16* p) {
  union { uint4 q[2]; v16bf v; } u;
  u.q[0] = *(const uint4*)p;
  u.q[1] = *(const uint4*)(p + 8);
  return u.v;
}

// ---------------- per-batch mean ----------------
__global__ void __launch_bounds__(1024) k_batch_mean(const float* __restrict__ x,
                                                     float* __restrict__ stats) {
  __shared__ float red[1024];
  int b = blockIdx.x, tid = threadIdx.x;
  const float* xb = x + (size_t)b * C_ * HW;
  float s = 0.f;
  for (int i = tid; i < C_ * HW; i += 1024) s += xb[i];
  red[tid] = s; __syncthreads();
  for (int st = 512; st > 0; st >>= 1) {
    if (tid < st) red[tid] += red[tid + st];
    __syncthreads();
  }
  if (tid == 0) stats[b * 8 + 0] = red[0] / (float)(C_ * HW);
}

// ---------------- relu-split-norm statistics ----------------
__global__ void __launch_bounds__(1024) k_split_stats(const float* __restrict__ x,
                                                      float* __restrict__ stats) {
  __shared__ float r0[1024], r1[1024], r2[1024], r3[1024];
  int b = blockIdx.x, tid = threadIdx.x;
  float m = stats[b * 8 + 0];
  const float* xb = x + (size_t)b * C_ * HW;
  float pos = 0.f, s1 = 0.f, sp2 = 0.f, sn2 = 0.f;
  for (int i = tid; i < C_ * HW; i += 1024) {
    float xc = xb[i] - m;
    if (xc > 0.f) { pos += 1.f; s1 += xc; sp2 += xc * xc; }
    else          { sn2 += xc * xc; }
  }
  r0[tid] = pos; r1[tid] = s1; r2[tid] = sp2; r3[tid] = sn2; __syncthreads();
  for (int st = 512; st > 0; st >>= 1) {
    if (tid < st) {
      r0[tid] += r0[tid + st]; r1[tid] += r1[tid + st];
      r2[tid] += r2[tid + st]; r3[tid] += r3[tid + st];
    }
    __syncthreads();
  }
  if (tid == 0) {
    float total = (float)(C_ * HW);
    pos = r0[0]; s1 = r1[0]; sp2 = r2[0]; sn2 = r3[0];
    float neg  = total - pos;
    float avg1 = s1 / pos;
    float var1 = (sp2 + avg1 * avg1 * neg) / total - avg1 * avg1;
    float k1   = sqrtf(pos / total) * rsqrtf(var1 + EPS_);
    float avg2 = -s1 / neg;
    float var2 = (sn2 + avg2 * avg2 * pos) / total - avg2 * avg2;
    float k2   = sqrtf(neg / total) * rsqrtf(var2 + EPS_);
    stats[b * 8 + 1] = avg1; stats[b * 8 + 2] = k1;
    stats[b * 8 + 3] = avg2; stats[b * 8 + 4] = k2;
  }
}

// ---------------- elementwise split: x1,x2 in bf16 (per-batch chunks) -------
__global__ void __launch_bounds__(256) k_elemwise_split(
    const float* __restrict__ x, const float* __restrict__ stats,
    const float* __restrict__ sc1p, const float* __restrict__ sc2p,
    bf16* __restrict__ x1bf, bf16* __restrict__ x2bf) {
  int b = blockIdx.x >> 6, chunk = blockIdx.x & 63;
  const int CH = C_ * HW / 64;  // 24576
  float sc1 = *sc1p, sc2 = *sc2p;
  float m = stats[b * 8], avg1 = stats[b * 8 + 1], k1 = stats[b * 8 + 2];
  float avg2 = stats[b * 8 + 3], k2 = stats[b * 8 + 4];
  size_t base = (size_t)b * C_ * HW + (size_t)chunk * CH;
  for (int i = threadIdx.x; i < CH; i += 256) {
    float xc = x[base + i] - m;
    float x1 = (xc > 0.f) ? sc1 * xc + k1 * (xc - avg1) : 0.f;
    float x2 = (xc > 0.f) ? 0.f : sc2 * xc + k2 * (xc - avg2);
    x1bf[base + i] = (bf16)x1;
    x2bf[base + i] = (bf16)x2;
  }
}

// ---------------- weight prepack into WMMA A-fragment order ----------------
// in: fp32 [nW][96][KS*32]  out: bf16 [nW][6][KS][32][16]
// perm=1: K' is kk-major (k' = kk*96 + ci  ->  src k = ci*9 + kk)
__global__ void __launch_bounds__(256) k_prepack(const float* __restrict__ w,
                                                 bf16* __restrict__ pk,
                                                 int KS, int nW, int perm) {
  int total = nW * 6 * KS * 512;
  int Ktot = KS * 32;
  for (int idx = blockIdx.x * blockDim.x + threadIdx.x; idx < total;
       idx += gridDim.x * blockDim.x) {
    int e = idx & 15, lane = (idx >> 4) & 31;
    int rest = idx >> 9;
    int ks = rest % KS; int mtw = rest / KS;
    int mt = mtw % 6, wm = mtw / 6;
    int co = mt * 16 + (lane & 15);
    int kp = ks * 32 + koffA(lane, e);
    int srck = kp;
    if (perm) { int kkid = kp / 96; int ci = kp - kkid * 96; srck = ci * 9 + kkid; }
    pk[idx] = (bf16)w[(size_t)wm * C_ * Ktot + (size_t)co * Ktot + srck];
  }
}

// ---------------- 3x3 conv as implicit GEMM (WMMA bf16, kk-major K) --------
// grid = B*H*2 ; block = 256 (8 waves). Tile: 96 out-ch x 64 pixels.
__global__ void __launch_bounds__(256) k_conv3x3(
    const bf16* __restrict__ in, const bf16* __restrict__ wpk, int wstride,
    float* __restrict__ outf, bf16* __restrict__ outbf) {
  __shared__ bf16 lds[3 * 66 * CST];  // [rr*66+wp][ci], 41184 B
  int bid = blockIdx.x;
  int tw = bid & 1;
  int h  = (bid >> 1) & 127;
  int b  = bid >> 8;
  int tid = threadIdx.x;

  // stage slab: coalesced global reads, transposed (ci-contiguous) LDS writes
  for (int idx = tid; idx < 3 * 96 * 66; idx += 256) {
    int wp = idx % 66; int t = idx / 66;
    int ci = t % 96;   int rr = t / 96;
    int r = h - 1 + rr;
    int w = tw * 64 + wp - 1;
    bf16 v = (bf16)0.f;
    if (r >= 0 && r < H_ && w >= 0 && w < W_)
      v = in[(((size_t)(b * C_ + ci)) << 14) + (r << 7) + w];
    lds[(rr * 66 + wp) * CST + ci] = v;
  }
  __syncthreads();

  int lane = tid & 31, wv = tid >> 5;
  int half = lane >> 4, ln = lane & 15;
  int ntile = wv & 3, mhalf = wv >> 2;
  int wcol = ntile * 16 + ln;
  const bf16* wb = wpk + (size_t)b * wstride +
                   (size_t)mhalf * 3 * KS3 * 512 + (size_t)lane * 16;
  __builtin_prefetch(wb, 0, 0);  // global_prefetch_b8 of A stream head

  v8f acc[3];
  #pragma unroll
  for (int mtl = 0; mtl < 3; ++mtl) acc[mtl] = v8f{0, 0, 0, 0, 0, 0, 0, 0};

  int bbase = wcol * CST + half * 16;  // + (kh*66+kw)*CST + cb*32 per step
  #pragma unroll
  for (int ks = 0; ks < KS3; ++ks) {
    int kkidx = ks / 3, cb = ks - kkidx * 3;       // compile-time constants
    int kh = kkidx / 3, kw = kkidx - kh * 3;
    v16bf bfr = ldfrag(&lds[bbase + (kh * 66 + kw) * CST + cb * 32]);
    #pragma unroll
    for (int mtl = 0; mtl < 3; ++mtl) {
      v16bf af = ldfrag(wb + (size_t)(mtl * KS3 + ks) * 512);
      acc[mtl] = __builtin_amdgcn_wmma_f32_16x16x32_bf16(
          false, af, false, bfr, (short)0, acc[mtl], false, false);
    }
  }

  #pragma unroll
  for (int mtl = 0; mtl < 3; ++mtl) {
    #pragma unroll
    for (int r = 0; r < 8; ++r) {
      int co = (mhalf * 3 + mtl) * 16 + r + half * 8;
      size_t o = (((size_t)(b * C_ + co)) << 14) + (h << 7) + tw * 64 + wcol;
      float v = acc[mtl][r];
      if (outf)  outf[o]  = v;
      if (outbf) outbf[o] = (bf16)v;
    }
  }
}

// -------- repack conv outputs into WMMA fragment streams for attention -----
// strm: [b][kk][tile 6][ks 56][lane 32][e 16]; zero-padded K to 1792
__global__ void __launch_bounds__(256) k_attn_repack(const bf16* __restrict__ img,
                                                     bf16* __restrict__ strm,
                                                     int isA) {
  for (long long idx = blockIdx.x * 256LL + threadIdx.x; idx < ASTRM;
       idx += (long long)gridDim.x * 256) {
    int e = (int)(idx & 15), lane = (int)((idx >> 4) & 31);
    int rest = (int)(idx >> 9);
    int ks = rest % AKS; int t = rest / AKS;
    int mt = t % 6; int t2 = t / 6;
    int kk = t2 % 9; int b = t2 / 9;
    int kh = kk / 3, kw = kk - kh * 3;
    int ch = mt * 16 + (lane & 15);
    int l  = isA ? (ks * 32 + koffA(lane, e))
                 : (ks * 32 + ((lane >> 4) << 4) + e);
    bf16 v = (bf16)0.f;
    if (l < LPATCH) {
      int i = l / 42, j = l - i * 42;
      v = img[(((size_t)(b * C_ + ch)) << 14) + ((3 * i + kh) << 7) + (3 * j + kw)];
    }
    strm[idx] = v;
  }
}

// ---------------- attention correlation GEMM (pure fragment loads) ---------
// grid = B*9 ; block = 256 (waves 0..5). M=96(q), N=96(c), K=1792.
__global__ void __launch_bounds__(256) k_attn_gemm(
    const bf16* __restrict__ qA, const bf16* __restrict__ kB,
    float* __restrict__ attn) {
  int bid = blockIdx.x;
  int b = bid / 9, kk = bid % 9;
  int tid = threadIdx.x, lane = tid & 31, wv = tid >> 5;
  if (wv >= 6) return;
  int half = lane >> 4, ln = lane & 15;

  size_t base = (size_t)(b * 9 + kk) * 6;
  const bf16* pb = kB + (base + wv) * AKS * 512 + (size_t)lane * 16;
  const bf16* pa = qA + base * AKS * 512 + (size_t)lane * 16;
  __builtin_prefetch(pb, 0, 0);
  __builtin_prefetch(pa, 0, 0);

  v8f acc[6];
  #pragma unroll
  for (int mt = 0; mt < 6; ++mt) acc[mt] = v8f{0, 0, 0, 0, 0, 0, 0, 0};

  #pragma unroll 2
  for (int ks = 0; ks < AKS; ++ks) {
    v16bf bfr = ldfrag(pb + (size_t)ks * 512);
    #pragma unroll
    for (int mt = 0; mt < 6; ++mt) {
      v16bf af = ldfrag(pa + (size_t)(mt * AKS + ks) * 512);
      acc[mt] = __builtin_amdgcn_wmma_f32_16x16x32_bf16(
          false, af, false, bfr, (short)0, acc[mt], false, false);
    }
  }
  #pragma unroll
  for (int mt = 0; mt < 6; ++mt) {
    #pragma unroll
    for (int r = 0; r < 8; ++r) {
      int q = mt * 16 + r + half * 8;
      int c = wv * 16 + ln;
      attn[((size_t)(b * C_ + q)) * KTOT3 + c * 9 + kk] = acc[mt][r];
    }
  }
}

// ---------------- softmax over 864, in place ----------------
__global__ void __launch_bounds__(256) k_softmax(float* __restrict__ attn) {
  __shared__ float red[256];
  int row = blockIdx.x, tid = threadIdx.x;
  float* a = attn + (size_t)row * KTOT3;
  const float sc = 0.0340206909f;  // 1/sqrt(864)
  float v[4]; float mx = -1e30f;
  #pragma unroll
  for (int t = 0; t < 4; ++t) {
    int i = tid + t * 256;
    v[t] = (i < KTOT3) ? a[i] * sc : -1e30f;
    mx = fmaxf(mx, v[t]);
  }
  red[tid] = mx; __syncthreads();
  for (int st = 128; st > 0; st >>= 1) {
    if (tid < st) red[tid] = fmaxf(red[tid], red[tid + st]);
    __syncthreads();
  }
  mx = red[0]; __syncthreads();
  float s = 0.f;
  #pragma unroll
  for (int t = 0; t < 4; ++t) { v[t] = __expf(v[t] - mx); s += v[t]; }
  red[tid] = s; __syncthreads();
  for (int st = 128; st > 0; st >>= 1) {
    if (tid < st) red[tid] += red[tid + st];
    __syncthreads();
  }
  float inv = 1.f / red[0];
  #pragma unroll
  for (int t = 0; t < 4; ++t) {
    int i = tid + t * 256;
    if (i < KTOT3) a[i] = v[t] * inv;
  }
}

// ---------------- per-batch layernorm stats (mu, 1/sigma) ----------------
__global__ void __launch_bounds__(1024) k_ln_stats(const float* __restrict__ x,
                                                   float* __restrict__ out2) {
  __shared__ float r0[1024], r1[1024];
  int b = blockIdx.x, tid = threadIdx.x;
  const float* xb = x + (size_t)b * C_ * HW;
  float s = 0.f, s2 = 0.f;
  for (int i = tid; i < C_ * HW; i += 1024) { float v = xb[i]; s += v; s2 += v * v; }
  r0[tid] = s; r1[tid] = s2; __syncthreads();
  for (int st = 512; st > 0; st >>= 1) {
    if (tid < st) { r0[tid] += r0[tid + st]; r1[tid] += r1[tid + st]; }
    __syncthreads();
  }
  if (tid == 0) {
    float N = (float)(C_ * HW);
    float mu = r0[0] / N;
    float var = r1[0] / N - mu * mu;
    out2[b * 2] = mu;
    out2[b * 2 + 1] = rsqrtf(var + EPS_);
  }
}

// ---------------- final fused: norms + gate + 1x1 WMMA conv + xr + bias ----
// grid = B*128 ; block = 256 (8 waves). Tile: 96 out x 128 pixels.
__global__ void __launch_bounds__(256) k_final(
    const float* __restrict__ x, const float* __restrict__ stats,
    const float* __restrict__ lnc2, const float* __restrict__ lna,
    const float* __restrict__ conv2f, const float* __restrict__ attnf,
    const bf16* __restrict__ x1bf, const bf16* __restrict__ pkfull,
    const float* __restrict__ bfull,
    const float* __restrict__ nc1p, const float* __restrict__ nc2p,
    const float* __restrict__ rcp, float* __restrict__ out) {
  __shared__ bf16 yl[128 * YST];  // [pixel][ci], 51200 B
  int bid = blockIdx.x;
  int b = bid >> 7, pt = bid & 127;
  int pbase = pt * 128;
  int tid = threadIdx.x;
  float mu_a = lna[b * 2],  rs_a = lna[b * 2 + 1];
  float mu_c = lnc2[b * 2], rs_c = lnc2[b * 2 + 1];

  for (int idx = tid; idx < 192 * 128; idx += 256) {
    int p = idx & 127, ci = idx >> 7;
    float y;
    if (ci < 96) {
      y = (attnf[(((size_t)(b * C_ + ci)) << 14) + pbase + p] - mu_a) * rs_a;
    } else {
      int c2 = ci - 96;
      size_t o = (((size_t)(b * C_ + c2)) << 14) + pbase + p;
      y = (conv2f[o] - mu_c) * rs_c * (float)x1bf[o];
    }
    yl[p * YST + ci] = (bf16)y;
  }
  __syncthreads();

  int lane = tid & 31, wv = tid >> 5;
  int half = lane >> 4, ln = lane & 15;
  int pcol = wv * 16 + ln;
  const bf16* pw = pkfull + (size_t)lane * 16;

  v8f acc[6];
  #pragma unroll
  for (int mt = 0; mt < 6; ++mt) acc[mt] = v8f{0, 0, 0, 0, 0, 0, 0, 0};

  #pragma unroll
  for (int ks = 0; ks < KS1; ++ks) {
    v16bf bfr = ldfrag(&yl[pcol * YST + ks * 32 + half * 16]);
    #pragma unroll
    for (int mt = 0; mt < 6; ++mt) {
      v16bf af = ldfrag(pw + (size_t)(mt * KS1 + ks) * 512);
      acc[mt] = __builtin_amdgcn_wmma_f32_16x16x32_bf16(
          false, af, false, bfr, (short)0, acc[mt], false, false);
    }
  }

  float nc1 = *nc1p, nc2 = *nc2p, rc = *rcp;
  float m = stats[b * 8], avg1 = stats[b * 8 + 1], k1 = stats[b * 8 + 2];
  float avg2 = stats[b * 8 + 3], k2 = stats[b * 8 + 4];
  #pragma unroll
  for (int mt = 0; mt < 6; ++mt) {
    #pragma unroll
    for (int r = 0; r < 8; ++r) {
      int co = mt * 16 + r + half * 8;
      size_t o = (((size_t)(b * C_ + co)) << 14) + pbase + pcol;
      float xc = x[o] - m;
      float xr = (xc > 0.f) ? 0.5f * nc1 * k1 * (xc - avg1) + rc * xc
                            : 0.5f * nc2 * k2 * (xc - avg2) + rc * xc;
      out[o] = acc[mt][r] + xr + bfull[co];
    }
  }
}

// ---------------- host orchestration ----------------
extern "C" void kernel_launch(void* const* d_in, const int* in_sizes, int n_in,
                              void* d_out, int out_size, void* d_ws, size_t ws_size,
                              hipStream_t stream) {
  (void)in_sizes; (void)n_in; (void)out_size;
  const float* x      = (const float*)d_in[0];
  const float* wconv2 = (const float*)d_in[1];
  const float* aw1    = (const float*)d_in[2];
  const float* aw2    = (const float*)d_in[3];
  const float* aw3    = (const float*)d_in[4];
  const float* wfull  = (const float*)d_in[5];
  const float* bfull  = (const float*)d_in[6];
  const float* sc1    = (const float*)d_in[7];
  const float* sc2    = (const float*)d_in[8];
  const float* rc     = (const float*)d_in[9];
  const float* nc1    = (const float*)d_in[10];
  const float* nc2    = (const float*)d_in[11];
  float* out = (float*)d_out;

  char* ws = (char*)d_ws;
  size_t off = 0;
  auto carve = [&](size_t bytes) -> char* {
    char* p = ws + off;
    off += (bytes + 255) & ~(size_t)255;
    return p;
  };
  float* stats  = (float*)carve(4096);
  float* lnc2   = (float*)carve(256);
  float* lna    = (float*)carve(256);
  bf16*  x1bf   = (bf16*)carve((size_t)NCHW * 2);
  bf16*  x2bf   = (bf16*)carve((size_t)NCHW * 2);
  bf16*  keyb   = (bf16*)carve((size_t)NCHW * 2);
  bf16*  qryb   = (bf16*)carve((size_t)NCHW * 2);
  bf16*  valb   = (bf16*)carve((size_t)NCHW * 2);
  float* conv2f = (float*)carve((size_t)NCHW * 4);
  float* attnof = (float*)carve((size_t)NCHW * 4);
  float* attnr  = (float*)carve((size_t)B_ * C_ * KTOT3 * 4);
  bf16*  qAs    = (bf16*)carve((size_t)ASTRM * 2);
  bf16*  kBs    = (bf16*)carve((size_t)ASTRM * 2);
  bf16*  pk1    = (bf16*)carve((size_t)PK3_SZ * 2);
  bf16*  pk2    = (bf16*)carve((size_t)PK3_SZ * 2);
  bf16*  pk3    = (bf16*)carve((size_t)PK3_SZ * 2);
  bf16*  pkc2   = (bf16*)carve((size_t)PK3_SZ * 2);
  bf16*  pkat   = (bf16*)carve((size_t)B_ * PK3_SZ * 2);
  bf16*  pkful  = (bf16*)carve((size_t)PK1_SZ * 2);
  if (off > ws_size) return;

  // 1) relu-split-norm statistics + elementwise split
  k_batch_mean<<<B_, 1024, 0, stream>>>(x, stats);
  k_split_stats<<<B_, 1024, 0, stream>>>(x, stats);
  k_elemwise_split<<<B_ * 64, 256, 0, stream>>>(x, stats, sc1, sc2, x1bf, x2bf);

  // 2) weight prepack (kk-major K for the 3x3 convs)
  int pgrid3 = (PK3_SZ + 255) / 256;
  k_prepack<<<pgrid3, 256, 0, stream>>>(aw1,    pk1,  KS3, 1, 1);
  k_prepack<<<pgrid3, 256, 0, stream>>>(aw2,    pk2,  KS3, 1, 1);
  k_prepack<<<pgrid3, 256, 0, stream>>>(aw3,    pk3,  KS3, 1, 1);
  k_prepack<<<pgrid3, 256, 0, stream>>>(wconv2, pkc2, KS3, 1, 1);
  k_prepack<<<(PK1_SZ + 255) / 256, 256, 0, stream>>>(wfull, pkful, KS1, 1, 0);

  // 3) four static 3x3 convs via WMMA implicit-GEMM
  int cgrid = B_ * H_ * 2;
  k_conv3x3<<<cgrid, 256, 0, stream>>>(x1bf, pk1,  0, nullptr, keyb);
  k_conv3x3<<<cgrid, 256, 0, stream>>>(x2bf, pk2,  0, nullptr, qryb);
  k_conv3x3<<<cgrid, 256, 0, stream>>>(x1bf, pk3,  0, nullptr, valb);
  k_conv3x3<<<cgrid, 256, 0, stream>>>(x2bf, pkc2, 0, conv2f, nullptr);

  // 4) attention: fragment-stream repack, correlation GEMMs, softmax, prepack
  k_attn_repack<<<4096, 256, 0, stream>>>(qryb, qAs, 1);
  k_attn_repack<<<4096, 256, 0, stream>>>(keyb, kBs, 0);
  k_attn_gemm<<<B_ * 9, 256, 0, stream>>>(qAs, kBs, attnr);
  k_softmax<<<B_ * C_, 256, 0, stream>>>(attnr);
  k_prepack<<<(B_ * PK3_SZ + 255) / 256, 256, 0, stream>>>(attnr, pkat, KS3, B_, 1);

  // 5) per-example attention conv (data-dependent weights)
  k_conv3x3<<<cgrid, 256, 0, stream>>>(valb, pkat, PK3_SZ, attnof, nullptr);

  // 6) layernorm stats for both branches
  k_ln_stats<<<B_, 1024, 0, stream>>>(conv2f, lnc2);
  k_ln_stats<<<B_, 1024, 0, stream>>>(attnof, lna);

  // 7) fused final: norms + gate + 1x1 WMMA conv + residual + bias
  k_final<<<B_ * 128, 256, 0, stream>>>(x, stats, lnc2, lna, conv2f, attnof,
                                        x1bf, pkful, bfull, nc1, nc2, rc, out);
}
